// FlashCRA_3118146257504
// MI455X (gfx1250) — matrix operand order repaired and checked by reading, not compiled, not run on hardware
//
#include <hip/hip_runtime.h>
#include <hip/hip_bf16.h>

// ---------------------------------------------------------------------------
// Types
// ---------------------------------------------------------------------------
typedef __bf16 bf16;
typedef __attribute__((ext_vector_type(16))) __bf16        v16bf;
typedef __attribute__((ext_vector_type(8)))  float         v8f;
typedef __attribute__((ext_vector_type(4)))  int           v4i;
typedef __attribute__((ext_vector_type(4)))  unsigned int  u32x4;
typedef __attribute__((ext_vector_type(8)))  int           i32x8;
typedef __attribute__((ext_vector_type(4)))  int           i32x4;

#if __has_builtin(__builtin_amdgcn_tensor_load_to_lds) && \
    __has_builtin(__builtin_amdgcn_s_wait_tensorcnt)
#define USE_TDM 1
#else
#define USE_TDM 0
#endif

struct BF16Frag { v4i lo, hi; };

__device__ __forceinline__ v16bf make_frag(v4i lo, v4i hi) {
    BF16Frag t{lo, hi};
    return __builtin_bit_cast(v16bf, t);
}

// A-fragment (16x32 bf16, M x K), source row-major [m][k], leading dim ld.
// Lanes 0-15: M=lane, K = {0..7, 16..23}; lanes 16-31: M=lane-16, K = {8..15, 24..31}.
__device__ __forceinline__ v16bf load_frag_a(const bf16* __restrict__ A, int ld) {
    const int lane = threadIdx.x & 31;
    const int row  = lane & 15;
    const int kofs = (lane >> 4) ? 8 : 0;
    const v4i* p0 = reinterpret_cast<const v4i*>(A + (size_t)row * ld + kofs);
    const v4i* p1 = reinterpret_cast<const v4i*>(A + (size_t)row * ld + kofs + 16);
    return make_frag(*p0, *p1);
}

// B-fragment (32x16 bf16, K x N), source is B^T row-major [n][k], leading dim ld.
// Lane holds N = lane&15; lanes 0-15: K = {0..7, 16..23}; lanes 16-31: K = {8..15, 24..31}.
__device__ __forceinline__ v16bf load_frag_b(const bf16* __restrict__ Bt, int ld) {
    const int lane = threadIdx.x & 31;
    const int n    = lane & 15;
    const int kofs = (lane >> 4) ? 8 : 0;
    const v4i* p = reinterpret_cast<const v4i*>(Bt + (size_t)n * ld + kofs);
    return make_frag(p[0], p[2]);
}

__device__ __forceinline__ v8f wmma_bf16(v16bf a, v16bf b, v8f c) {
    return __builtin_amdgcn_wmma_f32_16x16x32_bf16(
        false, a, false, b, (short)0, c, false, false);
}

// ---------------------------------------------------------------------------
// Tensor Data Mover: DMA a 2D bf16 tile (tileW x tileH, row stride = strideElems)
// from global memory into LDS (contiguous, row-major tileW-element rows).
// D# layout per CDNA5 ISA ch.8 (group0: count/lds/global/type, group1: dims).
// ---------------------------------------------------------------------------
#if USE_TDM
__device__ __forceinline__ void tdm_load_tile_2d(
    const bf16* gptr, unsigned ldsByteAddr,
    unsigned tensorW, unsigned tensorH,
    unsigned tileW, unsigned tileH, unsigned strideElems)
{
    const unsigned long long ga = (unsigned long long)(size_t)gptr;

    u32x4 g0;
    g0.x = 1u;                                        // count=1, user D#, no gather
    g0.y = ldsByteAddr;                               // lds_addr (bytes)
    g0.z = (unsigned)(ga & 0xffffffffu);              // global_addr[31:0]
    g0.w = (unsigned)((ga >> 32) & 0x01ffffffu)       // global_addr[56:32]
         | (2u << 30);                                // type = 2 ("image")

    i32x8 g1 = {};
    g1[0] = (int)(1u << 16);                          // data_size=1 (2 bytes), mask=0
    g1[1] = (int)((tensorW & 0xffffu) << 16);         // tensor_dim0[15:0]
    g1[2] = (int)((tensorW >> 16) | ((tensorH & 0xffffu) << 16)); // dim0 hi | dim1 lo
    g1[3] = (int)((tensorH >> 16) | (tileW << 16));   // dim1 hi | tile_dim0
    g1[4] = (int)(tileH & 0xffffu);                   // tile_dim1 (tile_dim2=0)
    g1[5] = (int)strideElems;                         // tensor_dim0_stride[31:0]
    g1[6] = 0;                                        // stride hi | dim1_stride lo
    g1[7] = 0;

    i32x4 z4 = {};
#if __clang_major__ >= 23
    i32x8 z8 = {};
    __builtin_amdgcn_tensor_load_to_lds(g0, g1, z4, z4, z8, 0);
#else
    __builtin_amdgcn_tensor_load_to_lds(g0, g1, z4, z4, 0);
#endif
}

__device__ __forceinline__ unsigned lds_byte_addr(const void* p) {
    // Generic pointers to LDS carry the LDS byte offset in the low 32 bits
    // (flat aperture mapping).
    return (unsigned)(size_t)p;
}
#endif

// ---------------------------------------------------------------------------
// Problem constants
// ---------------------------------------------------------------------------
constexpr int Bc = 2, Lc = 2048, Dc = 1024, Hc = 16, HDc = 64;
constexpr int BL = Bc * Lc;          // 4096

// ---------------------------------------------------------------------------
// f32 -> bf16 convert
// ---------------------------------------------------------------------------
__global__ void cvt_f32_bf16_kernel(const float* __restrict__ in,
                                    bf16* __restrict__ out, int n) {
    int i = blockIdx.x * blockDim.x + threadIdx.x;
    if (i < n) out[i] = (bf16)in[i];
}

// proj [D][D/2] (f32) -> projT [D/2][D] (bf16)
__global__ void transpose_cvt_kernel(const float* __restrict__ in,
                                     bf16* __restrict__ out) {
    int i = blockIdx.x * blockDim.x + threadIdx.x;   // over 512*1024
    if (i >= (Dc / 2) * Dc) return;
    int d = i & (Dc - 1);
    int m = i >> 10;
    out[(size_t)m * Dc + d] = (bf16)in[(size_t)d * (Dc / 2) + m];
}

// ---------------------------------------------------------------------------
// Tiled bf16 WMMA GEMM:  C[M x N] = A[M x K] * B[N x K]^T (+ bias[n])
// Block = 128 threads (4 waves), tile 64(M) x 64(N), K-step 32.
// B tile (64x32 bf16 = 4KB) staged in LDS by the Tensor Data Mover,
// double-buffered so the DMA for tile k+1 overlaps the WMMAs of tile k.
// ---------------------------------------------------------------------------
__global__ __launch_bounds__(128) void gemm_bf16_kernel(
    const bf16* __restrict__ A, const bf16* __restrict__ B,
    const float* __restrict__ bias, void* __restrict__ Cout,
    int M, int N, int K, int outBf16)
{
    __shared__ __align__(16) bf16 Bs[2][64 * 32];

    const int lane  = threadIdx.x & 31;
    const int wave  = threadIdx.x >> 5;
    const int nTile = blockIdx.x * 64;
    const int mTile = blockIdx.y * 64;
    const int mWave = mTile + wave * 16;

    v8f acc[4] = {};

#if USE_TDM
    const bool tdmWave = (threadIdx.x < 32);
    // Prologue: DMA tile 0 into buffer 0.
    if (tdmWave) {
        tdm_load_tile_2d(B + (size_t)nTile * K, lds_byte_addr(&Bs[0][0]),
                         (unsigned)K, (unsigned)N, 32u, 64u, (unsigned)K);
        __builtin_amdgcn_s_wait_tensorcnt(0);
    }
    __syncthreads();

    int buf = 0;
    for (int k0 = 0; k0 < K; k0 += 32) {
        // Kick off DMA for the next K-slab while we compute on this one.
        if (tdmWave && (k0 + 32 < K)) {
            tdm_load_tile_2d(B + (size_t)nTile * K + k0 + 32,
                             lds_byte_addr(&Bs[buf ^ 1][0]),
                             (unsigned)K, (unsigned)N, 32u, 64u, (unsigned)K);
        }

        __builtin_prefetch(A + (size_t)mWave * K + k0 + 64, 0, 0);

        const bf16* Bsb = &Bs[buf][0];
        v16bf af = load_frag_a(A + (size_t)mWave * K + k0, K);
        v16bf b0 = load_frag_b(Bsb + 0 * 16 * 32, 32);
        v16bf b1 = load_frag_b(Bsb + 1 * 16 * 32, 32);
        v16bf b2 = load_frag_b(Bsb + 2 * 16 * 32, 32);
        v16bf b3 = load_frag_b(Bsb + 3 * 16 * 32, 32);
        acc[0] = wmma_bf16(af, b0, acc[0]);
        acc[1] = wmma_bf16(af, b1, acc[1]);
        acc[2] = wmma_bf16(af, b2, acc[2]);
        acc[3] = wmma_bf16(af, b3, acc[3]);

        if (tdmWave) __builtin_amdgcn_s_wait_tensorcnt(0);
        __syncthreads();
        buf ^= 1;
    }
#else
    // Fallback: cooperative VGPR-bounce staging, single buffer.
    for (int k0 = 0; k0 < K; k0 += 32) {
        {
            const int t  = threadIdx.x;
            const int bn = t >> 1;
            const int bk = (t & 1) * 16;
            const v4i* gp = reinterpret_cast<const v4i*>(
                B + (size_t)(nTile + bn) * K + k0 + bk);
            v4i* lp = reinterpret_cast<v4i*>(&Bs[0][0] + bn * 32 + bk);
            lp[0] = gp[0];
            lp[1] = gp[1];
        }
        __syncthreads();

        const bf16* Bsb = &Bs[0][0];
        v16bf af = load_frag_a(A + (size_t)mWave * K + k0, K);
        v16bf b0 = load_frag_b(Bsb + 0 * 16 * 32, 32);
        v16bf b1 = load_frag_b(Bsb + 1 * 16 * 32, 32);
        v16bf b2 = load_frag_b(Bsb + 2 * 16 * 32, 32);
        v16bf b3 = load_frag_b(Bsb + 3 * 16 * 32, 32);
        acc[0] = wmma_bf16(af, b0, acc[0]);
        acc[1] = wmma_bf16(af, b1, acc[1]);
        acc[2] = wmma_bf16(af, b2, acc[2]);
        acc[3] = wmma_bf16(af, b3, acc[3]);
        __syncthreads();
    }
#endif

    // Epilogue. C layout: lane holds N = lane&15; VGPR v holds M = v + 8*(lane>>4).
    const int mBase = mWave + ((lane >> 4) ? 8 : 0);
    #pragma unroll
    for (int j = 0; j < 4; ++j) {
        const int n  = nTile + j * 16 + (lane & 15);
        const float bv = bias ? bias[n] : 0.0f;
        #pragma unroll
        for (int v = 0; v < 8; ++v) {
            const int m = mBase + v;
            const float val = acc[j][v] + bv;
            if (outBf16) ((bf16*)Cout)[(size_t)m * N + n] = (bf16)val;
            else         ((float*)Cout)[(size_t)m * N + n] = val;
        }
    }
}

// ---------------------------------------------------------------------------
// RoPE rotation + head-major split.
// ---------------------------------------------------------------------------
__global__ void rope_split_kernel(const bf16* __restrict__ qkv,
                                  const bf16* __restrict__ ph,
                                  bf16* __restrict__ Qo,
                                  bf16* __restrict__ Ko,
                                  bf16* __restrict__ Vt)
{
    int idx = blockIdx.x * blockDim.x + threadIdx.x;
    if (idx >= Bc * Lc * Hc * 32) return;
    const int d = idx & 31;
    const int h = (idx >> 5) & 15;
    const int l = (idx >> 9) & (Lc - 1);
    const int b = idx >> 20;

    const size_t row  = (size_t)b * Lc + l;
    const bf16* base  = qkv + row * (3 * Dc);
    const int   off   = h * HDc + d;

    float s, c;
    __sincosf((float)ph[row * (Dc / 2) + h * 32 + d], &s, &c);

    const float q1 = (float)base[off],            q2 = (float)base[off + 32];
    const float k1 = (float)base[Dc + off],       k2 = (float)base[Dc + off + 32];
    const float v1 = (float)base[2 * Dc + off],   v2 = (float)base[2 * Dc + off + 32];

    const size_t bh = (size_t)b * Hc + h;
    bf16* qp = Qo + (bh * Lc + l) * HDc;
    bf16* kp = Ko + (bh * Lc + l) * HDc;
    qp[d]      = (bf16)(q1 * c - q2 * s);
    qp[d + 32] = (bf16)(q1 * s + q2 * c);
    kp[d]      = (bf16)(k1 * c - k2 * s);
    kp[d + 32] = (bf16)(k1 * s + k2 * c);
    Vt[(bh * HDc + d)      * Lc + l] = (bf16)v1;
    Vt[(bh * HDc + d + 32) * Lc + l] = (bf16)v2;
}

// ---------------------------------------------------------------------------
// Flash attention: 1 wave per 16 queries per (b,h).
// ---------------------------------------------------------------------------
__global__ __launch_bounds__(32) void flash_attn_kernel(
    const bf16* __restrict__ Q, const bf16* __restrict__ Kk,
    const bf16* __restrict__ Vt, bf16* __restrict__ O)
{
    __shared__ __align__(16) bf16 Ps[16 * 32];

    const int lane = threadIdx.x & 31;
    const int q0   = blockIdx.x * 16;
    const int h    = blockIdx.y;
    const int b    = blockIdx.z;
    const size_t bh = (size_t)b * Hc + h;

    const bf16* Qb = Q  + (bh * Lc + q0) * HDc;
    const bf16* Kb = Kk +  bh * Lc * HDc;
    const bf16* Vb = Vt +  bh * HDc * Lc;

    const v16bf qa0 = load_frag_a(Qb, HDc);        // d = 0..31
    const v16bf qa1 = load_frag_a(Qb + 32, HDc);   // d = 32..63

    v8f acc[4] = {};
    float mi[8], li[8];
    #pragma unroll
    for (int v = 0; v < 8; ++v) { mi[v] = -1e30f; li[v] = 0.0f; }

    const float scale = 0.125f;   // HD^-0.5

    for (int k0 = 0; k0 < Lc; k0 += 32) {
        v8f s0 = {}, s1 = {};
        {
            v16bf kb = load_frag_b(Kb + (size_t)k0 * HDc, HDc);
            s0 = wmma_bf16(qa0, kb, s0);
            kb = load_frag_b(Kb + (size_t)k0 * HDc + 32, HDc);
            s0 = wmma_bf16(qa1, kb, s0);
        }
        {
            v16bf kb = load_frag_b(Kb + (size_t)(k0 + 16) * HDc, HDc);
            s1 = wmma_bf16(qa0, kb, s1);
            kb = load_frag_b(Kb + (size_t)(k0 + 16) * HDc + 32, HDc);
            s1 = wmma_bf16(qa1, kb, s1);
        }

        #pragma unroll
        for (int v = 0; v < 8; ++v) {
            float x0 = s0[v] * scale, x1 = s1[v] * scale;
            float mx = fmaxf(x0, x1);
            #pragma unroll
            for (int off = 8; off; off >>= 1)
                mx = fmaxf(mx, __shfl_xor(mx, off, 32));
            const float mnew  = fmaxf(mi[v], mx);
            const float alpha = __expf(mi[v] - mnew);
            const float p0 = __expf(x0 - mnew);
            const float p1 = __expf(x1 - mnew);
            float rs = p0 + p1;
            #pragma unroll
            for (int off = 8; off; off >>= 1)
                rs += __shfl_xor(rs, off, 32);
            li[v] = li[v] * alpha + rs;
            mi[v] = mnew;
            #pragma unroll
            for (int j = 0; j < 4; ++j) acc[j][v] *= alpha;

            const int m = v + ((lane >> 4) ? 8 : 0);
            Ps[m * 32 + (lane & 15)]      = (bf16)p0;
            Ps[m * 32 + 16 + (lane & 15)] = (bf16)p1;
        }
        __syncthreads();

        const v16bf pa = load_frag_a(Ps, 32);   // P: 16 q x 32 keys
        #pragma unroll
        for (int j = 0; j < 4; ++j) {
            v16bf vb = load_frag_b(Vb + (size_t)j * 16 * Lc + k0, Lc);
            acc[j] = wmma_bf16(pa, vb, acc[j]);
        }
        __syncthreads();
    }

    const int mHalf = (lane >> 4) ? 8 : 0;
    #pragma unroll
    for (int j = 0; j < 4; ++j) {
        #pragma unroll
        for (int v = 0; v < 8; ++v) {
            const int m = q0 + v + mHalf;
            const int n = h * HDc + j * 16 + (lane & 15);
            O[((size_t)b * Lc + m) * Dc + n] = (bf16)(acc[j][v] / li[v]);
        }
    }
}

// ---------------------------------------------------------------------------
// Host-side orchestration
// ---------------------------------------------------------------------------
extern "C" void kernel_launch(void* const* d_in, const int* in_sizes, int n_in,
                              void* d_out, int out_size, void* d_ws, size_t ws_size,
                              hipStream_t stream) {
    const float* x     = (const float*)d_in[0];
    const float* p     = (const float*)d_in[1];
    const float* W_qkv = (const float*)d_in[2];
    const float* b_qkv = (const float*)d_in[3];
    const float* W_out = (const float*)d_in[4];
    const float* b_out = (const float*)d_in[5];
    const float* proj  = (const float*)d_in[6];
    float* out = (float*)d_out;

    char* w = (char*)d_ws;
    auto alloc = [&](size_t bytes) -> char* {
        char* q = w;
        w += (bytes + 255) & ~(size_t)255;
        return q;
    };

    bf16* xb      = (bf16*)alloc((size_t)BL * Dc * 2);
    bf16* pb      = (bf16*)alloc((size_t)BL * Dc * 2);
    bf16* Wqkvb   = (bf16*)alloc((size_t)3 * Dc * Dc * 2);
    bf16* Woutb   = (bf16*)alloc((size_t)Dc * Dc * 2);
    bf16* projTb  = (bf16*)alloc((size_t)(Dc / 2) * Dc * 2);
    bf16* qkvb    = (bf16*)alloc((size_t)BL * 3 * Dc * 2);
    bf16* phasesb = (bf16*)alloc((size_t)BL * (Dc / 2) * 2);
    bf16* qrot    = (bf16*)alloc((size_t)Bc * Hc * Lc * HDc * 2);
    bf16* krot    = (bf16*)alloc((size_t)Bc * Hc * Lc * HDc * 2);
    bf16* vT      = (bf16*)alloc((size_t)Bc * Hc * HDc * Lc * 2);
    bf16* attnO   = (bf16*)alloc((size_t)BL * Dc * 2);

    const int cvtBlk = 256;
    cvt_f32_bf16_kernel<<<(BL * Dc + cvtBlk - 1) / cvtBlk, cvtBlk, 0, stream>>>(x, xb, BL * Dc);
    cvt_f32_bf16_kernel<<<(BL * Dc + cvtBlk - 1) / cvtBlk, cvtBlk, 0, stream>>>(p, pb, BL * Dc);
    cvt_f32_bf16_kernel<<<(3 * Dc * Dc + cvtBlk - 1) / cvtBlk, cvtBlk, 0, stream>>>(W_qkv, Wqkvb, 3 * Dc * Dc);
    cvt_f32_bf16_kernel<<<(Dc * Dc + cvtBlk - 1) / cvtBlk, cvtBlk, 0, stream>>>(W_out, Woutb, Dc * Dc);
    transpose_cvt_kernel<<<((Dc / 2) * Dc + cvtBlk - 1) / cvtBlk, cvtBlk, 0, stream>>>(proj, projTb);

    // phases = p @ proj   (M=4096, N=512, K=1024)
    gemm_bf16_kernel<<<dim3((Dc / 2) / 64, BL / 64), 128, 0, stream>>>(
        pb, projTb, nullptr, phasesb, BL, Dc / 2, Dc, 1);

    // qkv = x @ W_qkv^T + b_qkv   (M=4096, N=3072, K=1024)
    gemm_bf16_kernel<<<dim3(3 * Dc / 64, BL / 64), 128, 0, stream>>>(
        xb, Wqkvb, b_qkv, qkvb, BL, 3 * Dc, Dc, 1);

    // RoPE + split into head-major Q, K and transposed V
    {
        const int n = Bc * Lc * Hc * 32;
        rope_split_kernel<<<(n + cvtBlk - 1) / cvtBlk, cvtBlk, 0, stream>>>(
            qkvb, phasesb, qrot, krot, vT);
    }

    // Flash attention: grid (L/16, H, B), 1 wave per block
    flash_attn_kernel<<<dim3(Lc / 16, Hc, Bc), 32, 0, stream>>>(qrot, krot, vT, attnO);

    // out = attnO @ W_out^T + b_out   (M=4096, N=1024, K=1024), fp32 output
    gemm_bf16_kernel<<<dim3(Dc / 64, BL / 64), 128, 0, stream>>>(
        attnO, Woutb, b_out, out, BL, Dc, Dc, 0);
}
